// FSGNN_36953898615153
// MI455X (gfx1250) — matrix operand order repaired and checked by reading, not compiled, not run on hardware
//
#include <hip/hip_runtime.h>
#include <math.h>

typedef float  v2f   __attribute__((ext_vector_type(2)));
typedef float  v8f   __attribute__((ext_vector_type(8)));
typedef __bf16 v16bf __attribute__((ext_vector_type(16)));
typedef unsigned short u16;

#define N_     4096
#define NFEAT_ 1024
#define NHID_  512
#define NCLS_  32
#define YPADW  64

// bf16 round-to-nearest-even of a float, returned as 16-bit pattern
__device__ __forceinline__ unsigned bf16rne(float f) {
    unsigned u = __float_as_uint(f);
    return (u + 0x7fffu + ((u >> 16) & 1u)) >> 16;
}

// Async DMA: 16 bytes global -> LDS per lane, tracked by ASYNCcnt (no VGPR data path)
__device__ __forceinline__ void async_copy_b128(unsigned lds_off, const u16* g) {
    unsigned long long ga = (unsigned long long)(uintptr_t)g;
    asm volatile("global_load_async_to_lds_b128 %0, %1, off" :: "v"(lds_off), "v"(ga) : "memory");
}
__device__ __forceinline__ void wait_async_pending(int pending) {
    if (pending) asm volatile("s_wait_asynccnt 6" ::: "memory");
    else         asm volatile("s_wait_asynccnt 0" ::: "memory");
}

// =====================================================================
// Split-precision GEMM (Markidis, bf16 hi/lo planes, f32 accumulate):
//   C[M x Nc] = (AH+AL) @ (BH+BL) ~= AH@BH + AH@BL + AL@BH
// A planes row-major [M][K]; B planes stored TRANSPOSED [Nc][K] so every
// LDS tile is a verbatim block copy -> staged with GLOBAL_LOAD_ASYNC_TO_LDS_B128,
// double-buffered (ASYNCcnt pipeline), 12 V_WMMA_F32_16X16X32_BF16 per wave/stage.
// WRITE_S: 0 = f32 C only; 1 = + hi/lo planes of C; 2 = + transposed planes too.
// =====================================================================
template<int WRITE_S>
__global__ __launch_bounds__(256)
void gemm_split_wmma(const u16* __restrict__ AH, const u16* __restrict__ AL,
                     const u16* __restrict__ BHT, const u16* __restrict__ BLT,
                     float* __restrict__ C,
                     u16* __restrict__ SH, u16* __restrict__ SL,
                     u16* __restrict__ SHT, u16* __restrict__ SLT,
                     int K, int Nc, int lda, int ldb)
{
    __shared__ __align__(16) u16 AsH[2][128][40];  // 128x32 bf16, 80B rows
    __shared__ __align__(16) u16 AsL[2][128][40];
    __shared__ __align__(16) u16 BsHT[2][64][40];  // B^T tile: [col][K]
    __shared__ __align__(16) u16 BsLT[2][64][40];

    const int tid  = threadIdx.x;
    const int wave = tid >> 5;
    const int lane = tid & 31;
    const int l    = lane & 15;
    const int h    = lane >> 4;
    const int wm   = (wave >> 1) << 5;
    const int wn   = (wave & 1) << 5;
    const int m0   = blockIdx.y * 128;
    const int n0   = blockIdx.x * 64;

    v8f c00 = {}, c01 = {}, c10 = {}, c11 = {};
    union AFrag { v16bf v; int4 q[2]; };

    // 6 async b128 per thread per stage (4 A-plane + 2 B-plane slices)
    auto issue_stage = [&](int buf, int kk) {
        #pragma unroll
        for (int i = 0; i < 2; ++i) {
            int idx = tid + (i << 8);           // 0..511
            int r   = idx >> 2;                 // 0..127
            int c8  = (idx & 3) << 3;           // 0,8,16,24
            async_copy_b128((unsigned)(uintptr_t)&AsH[buf][r][c8],
                            AH + (size_t)(m0 + r) * lda + kk + c8);
            async_copy_b128((unsigned)(uintptr_t)&AsL[buf][r][c8],
                            AL + (size_t)(m0 + r) * lda + kk + c8);
        }
        {
            int r  = tid >> 2;                  // 0..63
            int c8 = (tid & 3) << 3;
            async_copy_b128((unsigned)(uintptr_t)&BsHT[buf][r][c8],
                            BHT + (size_t)(n0 + r) * ldb + kk + c8);
            async_copy_b128((unsigned)(uintptr_t)&BsLT[buf][r][c8],
                            BLT + (size_t)(n0 + r) * ldb + kk + c8);
        }
    };

    issue_stage(0, 0);
    if (32 < K) issue_stage(1, 32);

    for (int k0 = 0; k0 < K; k0 += 32) {
        const int buf = (k0 >> 5) & 1;
        wait_async_pending(k0 + 32 < K);   // in-order: current stage has landed
        __syncthreads();                   // all waves' slices visible

        // A frags: v0-3 = K[8h..8h+7], v4-7 = K[16+8h..]  (ISA 7.12.2)
        AFrag ah0, al0, ah1, al1;
        ah0.q[0] = *(const int4*)&AsH[buf][wm + l][8 * h];
        ah0.q[1] = *(const int4*)&AsH[buf][wm + l][16 + 8 * h];
        al0.q[0] = *(const int4*)&AsL[buf][wm + l][8 * h];
        al0.q[1] = *(const int4*)&AsL[buf][wm + l][16 + 8 * h];
        ah1.q[0] = *(const int4*)&AsH[buf][wm + 16 + l][8 * h];
        ah1.q[1] = *(const int4*)&AsH[buf][wm + 16 + l][16 + 8 * h];
        al1.q[0] = *(const int4*)&AsL[buf][wm + 16 + l][8 * h];
        al1.q[1] = *(const int4*)&AsL[buf][wm + 16 + l][16 + 8 * h];
        // B frags: v0-3 = K[8h..8h+7], v4-7 = K[16+8h..] (same order as A)
        AFrag bh0, bl0, bh1, bl1;
        bh0.q[0] = *(const int4*)&BsHT[buf][wn + l][8 * h];
        bh0.q[1] = *(const int4*)&BsHT[buf][wn + l][16 + 8 * h];
        bl0.q[0] = *(const int4*)&BsLT[buf][wn + l][8 * h];
        bl0.q[1] = *(const int4*)&BsLT[buf][wn + l][16 + 8 * h];
        bh1.q[0] = *(const int4*)&BsHT[buf][wn + 16 + l][8 * h];
        bh1.q[1] = *(const int4*)&BsHT[buf][wn + 16 + l][16 + 8 * h];
        bl1.q[0] = *(const int4*)&BsLT[buf][wn + 16 + l][8 * h];
        bl1.q[1] = *(const int4*)&BsLT[buf][wn + 16 + l][16 + 8 * h];

        c00 = __builtin_amdgcn_wmma_f32_16x16x32_bf16(false, ah0.v, false, bh0.v, (short)0, c00, false, false);
        c00 = __builtin_amdgcn_wmma_f32_16x16x32_bf16(false, ah0.v, false, bl0.v, (short)0, c00, false, false);
        c00 = __builtin_amdgcn_wmma_f32_16x16x32_bf16(false, al0.v, false, bh0.v, (short)0, c00, false, false);
        c01 = __builtin_amdgcn_wmma_f32_16x16x32_bf16(false, ah0.v, false, bh1.v, (short)0, c01, false, false);
        c01 = __builtin_amdgcn_wmma_f32_16x16x32_bf16(false, ah0.v, false, bl1.v, (short)0, c01, false, false);
        c01 = __builtin_amdgcn_wmma_f32_16x16x32_bf16(false, al0.v, false, bh1.v, (short)0, c01, false, false);
        c10 = __builtin_amdgcn_wmma_f32_16x16x32_bf16(false, ah1.v, false, bh0.v, (short)0, c10, false, false);
        c10 = __builtin_amdgcn_wmma_f32_16x16x32_bf16(false, ah1.v, false, bl0.v, (short)0, c10, false, false);
        c10 = __builtin_amdgcn_wmma_f32_16x16x32_bf16(false, al1.v, false, bh0.v, (short)0, c10, false, false);
        c11 = __builtin_amdgcn_wmma_f32_16x16x32_bf16(false, ah1.v, false, bh1.v, (short)0, c11, false, false);
        c11 = __builtin_amdgcn_wmma_f32_16x16x32_bf16(false, ah1.v, false, bl1.v, (short)0, c11, false, false);
        c11 = __builtin_amdgcn_wmma_f32_16x16x32_bf16(false, al1.v, false, bh1.v, (short)0, c11, false, false);

        __syncthreads();                   // everyone done reading this buffer
        if (k0 + 64 < K) issue_stage(buf, k0 + 64);   // refill just-freed buffer
    }

    // epilogue: C/D layout VGPR r -> M = r + 8h, N = l
    const int row0 = m0 + wm, col0 = n0 + wn;
    #pragma unroll
    for (int r = 0; r < 8; ++r) {
        float fs[4] = {c00[r], c01[r], c10[r], c11[r]};
        int   rr[4] = {row0 + r + 8 * h, row0 + r + 8 * h,
                       row0 + 16 + r + 8 * h, row0 + 16 + r + 8 * h};
        int   cc[4] = {col0 + l, col0 + 16 + l, col0 + l, col0 + 16 + l};
        #pragma unroll
        for (int t = 0; t < 4; ++t) {
            size_t idx = (size_t)rr[t] * Nc + cc[t];
            C[idx] = fs[t];
            if constexpr (WRITE_S >= 1) {
                unsigned hb = bf16rne(fs[t]);
                float hf = __uint_as_float(hb << 16);
                unsigned lb = bf16rne(fs[t] - hf);
                SH[idx] = (u16)hb;
                SL[idx] = (u16)lb;
                if constexpr (WRITE_S == 2) {
                    size_t ti = (size_t)cc[t] * N_ + rr[t];
                    SHT[ti] = (u16)hb;
                    SLT[ti] = (u16)lb;
                }
            }
        }
    }
}

// ---------------- f32 -> bf16 hi/lo planes (+ optional transposed planes) ----------------
__global__ void split_planes(const float* __restrict__ src,
                             u16* __restrict__ H, u16* __restrict__ L,
                             u16* __restrict__ HT, u16* __restrict__ LT,
                             int R, int Cc) {
    int idx = blockIdx.x * blockDim.x + threadIdx.x;
    if (idx >= R * Cc) return;
    int r = idx / Cc, c = idx % Cc;
    float f = src[idx];
    unsigned hb = bf16rne(f);
    float hf = __uint_as_float(hb << 16);
    unsigned lb = bf16rne(f - hf);
    if (H)  H[idx] = (u16)hb;
    if (L)  L[idx] = (u16)lb;
    if (HT) HT[(size_t)c * R + r] = (u16)hb;
    if (LT) LT[(size_t)c * R + r] = (u16)lb;
}

// =====================================================================
// f32 GEMM (V_WMMA_F32_16X16X4_F32) for the small matmuls.
//   BINA -> binarize A (x>0?1:0);  TRANSA -> A stored [K x M]
// =====================================================================
template<int BINA, int TRANSA>
__global__ __launch_bounds__(256)
void gemm_f32_wmma(const float* __restrict__ A, const float* __restrict__ B,
                   float* __restrict__ C, int K, int Nc, int lda)
{
    __shared__ __align__(16) float As[128][20];
    __shared__ __align__(16) float Bs[16][68];

    const int tid  = threadIdx.x;
    const int wave = tid >> 5;
    const int lane = tid & 31;
    const int l    = lane & 15;
    const int half = lane >> 4;
    const int wm   = (wave >> 1) << 5;
    const int wn   = (wave & 1) << 5;
    const int m0   = blockIdx.y * 128;
    const int n0   = blockIdx.x * 64;

    v8f c00 = {}, c01 = {}, c10 = {}, c11 = {};

    for (int k0 = 0; k0 < K; k0 += 16) {
        if (TRANSA) {
            #pragma unroll
            for (int i = 0; i < 8; ++i) {
                int idx = tid + (i << 8);
                int r = idx & 127;
                int c = idx >> 7;
                float v = A[(size_t)(k0 + c) * lda + (m0 + r)];
                if (BINA) v = (v > 0.0f) ? 1.0f : 0.0f;
                As[r][c] = v;
            }
        } else {
            #pragma unroll
            for (int i = 0; i < 2; ++i) {
                int idx = tid + (i << 8);
                int r  = idx >> 2;
                int c4 = (idx & 3) << 2;
                float4 v = *(const float4*)&A[(size_t)(m0 + r) * lda + k0 + c4];
                if (BINA) {
                    v.x = (v.x > 0.0f) ? 1.0f : 0.0f;
                    v.y = (v.y > 0.0f) ? 1.0f : 0.0f;
                    v.z = (v.z > 0.0f) ? 1.0f : 0.0f;
                    v.w = (v.w > 0.0f) ? 1.0f : 0.0f;
                }
                *(float4*)&As[r][c4] = v;
            }
        }
        {
            int r  = tid >> 4;
            int c4 = (tid & 15) << 2;
            float4 v = *(const float4*)&B[(size_t)(k0 + r) * Nc + n0 + c4];
            *(float4*)&Bs[r][c4] = v;
        }
        __syncthreads();

        #pragma unroll
        for (int kk = 0; kk < 16; kk += 4) {
            v2f a0 = *(const v2f*)&As[wm      + l][kk + 2 * half];
            v2f a1 = *(const v2f*)&As[wm + 16 + l][kk + 2 * half];
            v2f b0, b1;
            b0.x = Bs[kk + 2 * half    ][wn + l];
            b0.y = Bs[kk + 2 * half + 1][wn + l];
            b1.x = Bs[kk + 2 * half    ][wn + 16 + l];
            b1.y = Bs[kk + 2 * half + 1][wn + 16 + l];
            c00 = __builtin_amdgcn_wmma_f32_16x16x4_f32(false, a0, false, b0, (short)0, c00, false, false);
            c01 = __builtin_amdgcn_wmma_f32_16x16x4_f32(false, a0, false, b1, (short)0, c01, false, false);
            c10 = __builtin_amdgcn_wmma_f32_16x16x4_f32(false, a1, false, b0, (short)0, c10, false, false);
            c11 = __builtin_amdgcn_wmma_f32_16x16x4_f32(false, a1, false, b1, (short)0, c11, false, false);
        }
        __syncthreads();
    }

    const int row0 = m0 + wm;
    const int col0 = n0 + wn;
    #pragma unroll
    for (int r = 0; r < 8; ++r) {
        C[(size_t)(row0      + r + 8 * half) * Nc + col0      + l] = c00[r];
        C[(size_t)(row0      + r + 8 * half) * Nc + col0 + 16 + l] = c01[r];
        C[(size_t)(row0 + 16 + r + 8 * half) * Nc + col0      + l] = c10[r];
        C[(size_t)(row0 + 16 + r + 8 * half) * Nc + col0 + 16 + l] = c11[r];
    }
}

// ---------------- softmax over the 3x5 alpha table ----------------
__global__ void softmax3x5(const float* __restrict__ a, float* __restrict__ out) {
    if (threadIdx.x == 0) {
        for (int i = 0; i < 3; ++i) {
            float m = a[i * 5];
            for (int j = 1; j < 5; ++j) m = fmaxf(m, a[i * 5 + j]);
            float e[5], s = 0.0f;
            for (int j = 0; j < 5; ++j) { e[j] = __expf(a[i * 5 + j] - m); s += e[j]; }
            for (int j = 0; j < 5; ++j) out[i * 5 + j] = e[j] / s;
        }
    }
}

__global__ void pad_cols(const float* __restrict__ src, float* __restrict__ dst,
                         int rows, int cin, int cout) {
    int idx = blockIdx.x * blockDim.x + threadIdx.x;
    if (idx >= rows * cout) return;
    int r = idx / cout, c = idx % cout;
    dst[idx] = (c < cin) ? src[(size_t)r * cin + c] : 0.0f;
}

__global__ void build_mat(const float* __restrict__ hop, const float* __restrict__ yout,
                          const float* __restrict__ yin, const float* __restrict__ al,
                          float* __restrict__ mat, int d, int DW) {
    int idx = blockIdx.x * blockDim.x + threadIdx.x;
    if (idx >= N_ * DW) return;
    int r = idx / DW, c = idx % DW;
    float v;
    if (c < d) v = al[0] * hop[(size_t)r * d + c];
    else {
        int cc = c - d;
        if (cc < 32)      v = al[1] * yout[(size_t)r * YPADW + cc];
        else if (cc < 64) v = al[2] * yin [(size_t)r * YPADW + cc - 32];
        else              v = al[3] * (yin[(size_t)r * YPADW + cc - 64] +
                                       yout[(size_t)r * YPADW + cc - 64]);
    }
    mat[idx] = v;
}

__global__ __launch_bounds__(256)
void bias_l2norm_max(const float* __restrict__ lin, const float* __restrict__ bias,
                     const int* __restrict__ layer_norm, float* __restrict__ hid,
                     float* __restrict__ maxf, int first) {
    __shared__ float red[8];
    const int row = blockIdx.x, tid = threadIdx.x;
    float t0 = lin[(size_t)row * NHID_ + tid]       + bias[tid];
    float t1 = lin[(size_t)row * NHID_ + 256 + tid] + bias[256 + tid];
    float ss = t0 * t0 + t1 * t1;
    for (int o = 16; o > 0; o >>= 1) ss += __shfl_xor(ss, o, 32);
    if ((tid & 31) == 0) red[tid >> 5] = ss;
    __syncthreads();
    float tot = 0.0f;
    for (int i = 0; i < 8; ++i) tot += red[i];
    if (*layer_norm) {
        float dnm = fmaxf(sqrtf(tot), 1e-12f);
        t0 /= dnm; t1 /= dnm;
    }
    size_t i0 = (size_t)row * NHID_ + tid, i1 = i0 + 256;
    hid[i0] = t0; hid[i1] = t1;
    maxf[i0] = first ? t0 : fmaxf(maxf[i0], t0);
    maxf[i1] = first ? t1 : fmaxf(maxf[i1], t1);
}

__global__ void build_final(const float* __restrict__ features, const float* __restrict__ maxf,
                            float* __restrict__ fb) {
    int idx = blockIdx.x * blockDim.x + threadIdx.x;
    if (idx >= N_ * (NFEAT_ + NHID_)) return;
    int r = idx / (NFEAT_ + NHID_), c = idx % (NFEAT_ + NHID_);
    float v = (c < NFEAT_) ? features[(size_t)r * NFEAT_ + c]
                           : maxf[(size_t)r * NHID_ + (c - NFEAT_)];
    fb[idx] = fmaxf(v, 0.0f);
}

__global__ __launch_bounds__(256)
void log_softmax32(const float* __restrict__ logits, const float* __restrict__ b2,
                   float* __restrict__ out) {
    int row  = blockIdx.x * 8 + (threadIdx.x >> 5);
    int lane = threadIdx.x & 31;
    float x = logits[(size_t)row * YPADW + lane] + b2[lane];
    float m = x;
    for (int o = 16; o > 0; o >>= 1) m = fmaxf(m, __shfl_xor(m, o, 32));
    float e = __expf(x - m), s = e;
    for (int o = 16; o > 0; o >>= 1) s += __shfl_xor(s, o, 32);
    out[(size_t)row * NCLS_ + lane] = x - m - __logf(s);
}

// =====================================================================
extern "C" void kernel_launch(void* const* d_in, const int* in_sizes, int n_in,
                              void* d_out, int out_size, void* d_ws, size_t ws_size,
                              hipStream_t stream)
{
    const float* adj        = (const float*)d_in[0];
    const float* features   = (const float*)d_in[1];
    const float* y          = (const float*)d_in[2];
    const float* Wl[3]      = {(const float*)d_in[3], (const float*)d_in[5], (const float*)d_in[7]};
    const float* bl[3]      = {(const float*)d_in[4], (const float*)d_in[6], (const float*)d_in[8]};
    const float* W2         = (const float*)d_in[9];
    const float* b2         = (const float*)d_in[10];
    const float* alphas     = (const float*)d_in[11];
    const int*   layer_norm = (const int*)d_in[12];

    char* ws = (char*)d_ws;
    size_t off = 0;
    const size_t PLANE = (size_t)N_ * N_ * 2;                      // one bf16 plane
    float* padjF = (float*)(ws + off); off += (size_t)N_ * N_ * 4; // current power, f32
    u16* adjH  = (u16*)(ws + off); off += PLANE;
    u16* adjL  = (u16*)(ws + off); off += PLANE;
    u16* adjHT = (u16*)(ws + off); off += PLANE;
    u16* adjLT = (u16*)(ws + off); off += PLANE;
    u16* pH    = (u16*)(ws + off); off += PLANE;                   // adj^2 planes
    u16* pL    = (u16*)(ws + off); off += PLANE;
    u16* pHT   = (u16*)(ws + off); off += PLANE;
    u16* pLT   = (u16*)(ws + off); off += PLANE;
    u16* qH    = (u16*)(ws + off); off += PLANE;                   // adj^4 planes
    u16* qL    = (u16*)(ws + off); off += PLANE;
    u16* featHT = (u16*)(ws + off); off += (size_t)N_ * NFEAT_ * 2;
    u16* featLT = (u16*)(ws + off); off += (size_t)N_ * NFEAT_ * 2;
    float* hop  = (float*)(ws + off); off += (size_t)N_ * NFEAT_ * 4;
    float* mat  = (float*)(ws + off); off += (size_t)N_ * (NFEAT_ + 3 * NCLS_) * 4;
    float* lin  = (float*)(ws + off); off += (size_t)N_ * NHID_ * 4;
    float* hid  = (float*)(ws + off); off += (size_t)N_ * NHID_ * 4;
    float* maxf = (float*)(ws + off); off += (size_t)N_ * NHID_ * 4;
    float* ypad = (float*)(ws + off); off += (size_t)N_ * YPADW * 4;
    float* yout = (float*)(ws + off); off += (size_t)N_ * YPADW * 4;
    float* yin  = (float*)(ws + off); off += (size_t)N_ * YPADW * 4;
    float* W2p  = (float*)(ws + off); off += (size_t)(NFEAT_ + NHID_) * YPADW * 4;
    float* alph = (float*)(ws + off); off += 64;
    float* finalb = (float*)pH;                      // p planes dead after power chain
    float* logits = (float*)pH + (size_t)N_ * (NFEAT_ + NHID_);

    dim3 blk(256);
    const dim3 gAdjS(N_ / 64, N_ / 128);

    softmax3x5<<<1, 32, 0, stream>>>(alphas, alph);
    pad_cols<<<(N_ * YPADW + 255) / 256, blk, 0, stream>>>(y, ypad, N_, NCLS_, YPADW);
    pad_cols<<<((NFEAT_ + NHID_) * YPADW + 255) / 256, blk, 0, stream>>>(W2, W2p, NFEAT_ + NHID_, NCLS_, YPADW);
    // adj used as both A (row planes) and B (transposed planes)
    split_planes<<<(N_ * N_ + 255) / 256, blk, 0, stream>>>(adj, adjH, adjL, adjHT, adjLT, N_, N_);

    const float* padjf = adj;      // f32 view of current power (for binarize)
    const u16*   pAH = adjH; const u16* pAL = adjL;   // split view, A-operand form
    const float* feats = features;
    int d = NFEAT_;

    for (int i = 0; i < 3; ++i) {
        if (i == 1) {        // adj^2 = adj @ adj   (writes all four output planes)
            gemm_split_wmma<2><<<gAdjS, blk, 0, stream>>>(adjH, adjL, adjHT, adjLT, padjF,
                                                          pH, pL, pHT, pLT, N_, N_, N_, N_);
            padjf = padjF; pAH = pH; pAL = pL;
        } else if (i == 2) { // adj^4 = (adj^2) @ (adj^2)   (A-form planes only)
            gemm_split_wmma<1><<<gAdjS, blk, 0, stream>>>(pH, pL, pHT, pLT, padjF,
                                                          qH, qL, qH, qL, N_, N_, N_, N_);
            padjf = padjF; pAH = qH; pAL = qL;
        }
        // hop = pow_adj @ feats   (feats split into transposed B planes)
        split_planes<<<(N_ * d + 255) / 256, blk, 0, stream>>>(feats, (u16*)nullptr, (u16*)nullptr,
                                                               featHT, featLT, N_, d);
        gemm_split_wmma<0><<<dim3(d / 64, N_ / 128), blk, 0, stream>>>(pAH, pAL, featHT, featLT, hop,
                                                                       nullptr, nullptr, nullptr, nullptr,
                                                                       N_, d, N_, N_);
        // y_out = bin(pow_adj) @ y ;  y_in = bin(pow_adj)^T @ y   (f32 WMMA, sign-exact)
        gemm_f32_wmma<1, 0><<<dim3(YPADW / 64, N_ / 128), blk, 0, stream>>>(padjf, ypad, yout, N_, YPADW, N_);
        gemm_f32_wmma<1, 1><<<dim3(YPADW / 64, N_ / 128), blk, 0, stream>>>(padjf, ypad, yin,  N_, YPADW, N_);
        int DW = d + 3 * NCLS_;            // 1120 or 608, multiples of 16
        build_mat<<<(N_ * DW + 255) / 256, blk, 0, stream>>>(hop, yout, yin, alph + i * 5, mat, d, DW);
        gemm_f32_wmma<0, 0><<<dim3(NHID_ / 64, N_ / 128), blk, 0, stream>>>(mat, Wl[i], lin, DW, NHID_, DW);
        bias_l2norm_max<<<N_, blk, 0, stream>>>(lin, bl[i], layer_norm, hid, maxf, (i == 0) ? 1 : 0);
        feats = hid; d = NHID_;
    }

    build_final<<<(N_ * (NFEAT_ + NHID_) + 255) / 256, blk, 0, stream>>>(features, maxf, finalb);
    gemm_f32_wmma<0, 0><<<dim3(YPADW / 64, N_ / 128), blk, 0, stream>>>(finalb, W2p, logits,
                                                                        NFEAT_ + NHID_, YPADW, NFEAT_ + NHID_);
    log_softmax32<<<N_ / 8, blk, 0, stream>>>(logits, b2, (float*)d_out);

    (void)in_sizes; (void)n_in; (void)out_size; (void)ws_size;
}
